// AttentionCopyCoverageDecoder_35373350650621
// MI455X (gfx1250) — matrix operand order repaired and checked by reading, not compiled, not run on hardware
//
#include <hip/hip_runtime.h>

// Sizes (fixed by the reference)
#define B_   64
#define T_   400
#define E_   128
#define H2_  512
#define H4_  1024
#define V_   50000
#define G3_  1536   // 3*H2

typedef __attribute__((ext_vector_type(16))) __bf16 v16bf;
typedef __attribute__((ext_vector_type(8)))  float  v8f;

union BF16x16 { unsigned u[8]; v16bf v; };

// ---------------------------------------------------------------------------
// Packed f32x2 -> bf16x2 conversion (round-to-nearest-even in HW).
// ---------------------------------------------------------------------------
__device__ __forceinline__ unsigned cvt_pk_bf16(float lo, float hi) {
  unsigned d;
  asm("v_cvt_pk_bf16_f32 %0, %1, %2" : "=v"(d) : "v"(lo), "v"(hi));
  return d;
}

__device__ __forceinline__ float sigmoidf_(float x) {
  return 1.0f / (1.0f + __expf(-x));
}

// ---------------------------------------------------------------------------
// Block reductions (blockDim.x must be a power of two)
// ---------------------------------------------------------------------------
__device__ __forceinline__ float block_reduce_sum(float v, float* red) {
  int tid = threadIdx.x;
  red[tid] = v; __syncthreads();
  for (int s = blockDim.x >> 1; s > 0; s >>= 1) {
    if (tid < s) red[tid] += red[tid + s];
    __syncthreads();
  }
  float r = red[0]; __syncthreads();
  return r;
}
__device__ __forceinline__ float block_reduce_max(float v, float* red) {
  int tid = threadIdx.x;
  red[tid] = v; __syncthreads();
  for (int s = blockDim.x >> 1; s > 0; s >>= 1) {
    if (tid < s) red[tid] = fmaxf(red[tid], red[tid + s]);
    __syncthreads();
  }
  float r = red[0]; __syncthreads();
  return r;
}

// ---------------------------------------------------------------------------
// WMMA GEMM:  C[64,N] = act( A[64,K] @ W[N,K]^T + bias[N] )
// - One wave owns one 16-col tile and all four 16-row tiles (M=64), so every
//   W element is streamed from HBM exactly once (fc2_W = 204.8 MB dominates).
// - A is staged per-block into LDS as bf16 in K-chunks of 256 (32 KB): the
//   activations are converted once per block; inner-loop A fragments are
//   16B-aligned ds_load_b128, all EIGHT issued before the four WMMAs so the
//   dscnt waits are progressive instead of blocking before every WMMA.
// - W fragments: 16 contiguous f32 per lane (4x global_load_b128) + packed
//   v_cvt_pk_bf16_f32; global_prefetch of the next K-slab.
// Fragment layout used here (contiguous k) (16-bit A 16x32 / B 32x16, f32 C 16x16):
//   lane<16:  A K 0..15,  B K rows 0..15,  C rows M=0..7
//   lane>=16: A K 16..31, B K rows 16..31, C rows M=8..15
// ---------------------------------------------------------------------------
template <int ACT>   // 0 = none, 1 = tanh
__global__ void __launch_bounds__(256)
wmma_gemm_xwt(const float* __restrict__ A,    // [64, K]
              const float* __restrict__ W,    // [N, K] row-major
              const float* __restrict__ bias, // [N]
              float* __restrict__ C,          // [64, N]
              int N, int K)
{
  __shared__ __bf16 As[64 * 256];              // 32 KB A chunk (bf16)

  const int lane = threadIdx.x & 31;
  const int wave = threadIdx.x >> 5;
  const int nt   = blockIdx.x * 8 + wave;
  const int n0   = nt * 16;
  const bool active = (n0 < N);                // tail waves still hit barriers

  const int r16 = lane & 15;   // row within A tile / col within B,C tile
  const int hi  = lane >> 4;
  const int kbA = hi * 16;
  const int kbB = hi * 16;

  v8f acc[4] = {};
  const float* Wrow = active ? (W + (size_t)(n0 + r16) * K) : W;

  const int kchunk = (K < 256) ? K : 256;      // K is 128/512/1024
  const int ksh    = (kchunk == 256) ? 8 : 7;
  const int kmask  = kchunk - 1;

  for (int kc = 0; kc < K; kc += kchunk) {
    // ---- cooperative stage of A[:, kc:kc+kchunk] -> LDS bf16 ----
    for (int idx = threadIdx.x * 4; idx < (64 << ksh); idx += 1024) {
      const int row = idx >> ksh;
      const int kk  = idx & kmask;
      float4 a = *(const float4*)(A + (size_t)row * K + kc + kk);
      unsigned p0 = cvt_pk_bf16(a.x, a.y);
      unsigned p1 = cvt_pk_bf16(a.z, a.w);
      *(uint2*)(As + row * 256 + kk) = make_uint2(p0, p1);
    }
    __syncthreads();

    if (active) {
      for (int k0 = 0; k0 < kchunk; k0 += 32) {
        const int kg = kc + k0;
        if (kg + 32 < K) __builtin_prefetch(Wrow + kg + 32, 0, 1);

        // B fragment: W[(n0+r16), kg+kbB .. +15]  (contiguous)
        const float4* wp = (const float4*)(Wrow + kg + kbB);
        float4 w0 = wp[0], w1 = wp[1], w2 = wp[2], w3 = wp[3];
        BF16x16 bf;
        bf.u[0] = cvt_pk_bf16(w0.x, w0.y); bf.u[1] = cvt_pk_bf16(w0.z, w0.w);
        bf.u[2] = cvt_pk_bf16(w1.x, w1.y); bf.u[3] = cvt_pk_bf16(w1.z, w1.w);
        bf.u[4] = cvt_pk_bf16(w2.x, w2.y); bf.u[5] = cvt_pk_bf16(w2.z, w2.w);
        bf.u[6] = cvt_pk_bf16(w3.x, w3.y); bf.u[7] = cvt_pk_bf16(w3.z, w3.w);

        // Issue ALL A-fragment LDS loads first, then the 4 WMMAs, so the
        // ds latency overlaps matrix-op issue (progressive dscnt waits).
        BF16x16 af[4];
#pragma unroll
        for (int mt = 0; mt < 4; ++mt) {
          const __bf16* Ar = As + (mt * 16 + r16) * 256 + k0;
          uint4 lo4 = *(const uint4*)(Ar + kbA);        // elems 0..7
          uint4 hi4 = *(const uint4*)(Ar + 8 + kbA);    // elems 8..15
          af[mt].u[0]=lo4.x; af[mt].u[1]=lo4.y; af[mt].u[2]=lo4.z; af[mt].u[3]=lo4.w;
          af[mt].u[4]=hi4.x; af[mt].u[5]=hi4.y; af[mt].u[6]=hi4.z; af[mt].u[7]=hi4.w;
        }
#pragma unroll
        for (int mt = 0; mt < 4; ++mt) {
          acc[mt] = __builtin_amdgcn_wmma_f32_16x16x32_bf16(
              false, af[mt].v, false, bf.v, (short)0, acc[mt], false, false);
        }
      }
    }
    __syncthreads();
  }

  if (!active) return;
  const float bv = bias[n0 + r16];
#pragma unroll
  for (int mt = 0; mt < 4; ++mt) {
#pragma unroll
    for (int r = 0; r < 8; ++r) {
      float v = acc[mt][r] + bv;
      if (ACT == 1) v = tanhf(v);
      const int m = mt * 16 + hi * 8 + r;   // C layout: lane<16 M=r, else M=8+r
      C[(size_t)m * N + n0 + r16] = v;
    }
  }
}

// ---------------------------------------------------------------------------
// Small helper kernels
// ---------------------------------------------------------------------------
__global__ void zero_kernel(float* __restrict__ p, size_t n) {
  size_t i = (size_t)blockIdx.x * blockDim.x + threadIdx.x;
  if (i < n) p[i] = 0.0f;
}

__global__ void embed_kernel(const int* __restrict__ ids,
                             const float* __restrict__ table,
                             float* __restrict__ emb) {
  int b = blockIdx.x, e = threadIdx.x;           // 64 blocks x 128 threads
  emb[b * E_ + e] = table[(size_t)ids[b] * E_ + e];
}

// GRU gating: gi,gh are [B,1536]; h = (1-z)*n + z*h_prev
__global__ void gru_gate(const float* __restrict__ gi,
                         const float* __restrict__ gh,
                         const float* __restrict__ hprev,
                         float* __restrict__ hout) {
  int i = blockIdx.x * blockDim.x + threadIdx.x;   // B*H2 = 32768
  if (i >= B_ * H2_) return;
  int b = i / H2_, j = i - b * H2_;
  const float* gib = gi + (size_t)b * G3_;
  const float* ghb = gh + (size_t)b * G3_;
  float r = sigmoidf_(gib[j]            + ghb[j]);
  float z = sigmoidf_(gib[H2_ + j]      + ghb[H2_ + j]);
  float n = tanhf   (gib[2 * H2_ + j]   + r * ghb[2 * H2_ + j]);
  hout[i] = (1.0f - z) * n + z * hprev[i];
}

// Fused attention: scores (decomposed dot) -> tanh -> softmax(T) ->
// attn, new_coverage, coverage_loss (atomic), copy_prob scatter (atomic).
// The 2KB wE vector is staged into LDS via the gfx1250 async-to-LDS path.
__global__ void __launch_bounds__(256)
attention_kernel(const float* __restrict__ enc,     // [B,T,H2]
                 const float* __restrict__ h1,      // [B,H2]
                 const float* __restrict__ cov_in,  // [B,T]
                 const float* __restrict__ alignW,  // [1536]
                 const float* __restrict__ alignB,  // [1]
                 const float* __restrict__ covW,    // [H2]
                 const float* __restrict__ covB,    // [H2]
                 const int*   __restrict__ src,     // [B,T]
                 float* __restrict__ attn_out,      // [B,T]
                 float* __restrict__ new_cov,       // [B,T]
                 float* __restrict__ cov_loss,      // [1], pre-zeroed
                 float* __restrict__ copy_prob)     // [B,V], pre-zeroed
{
  const int b = blockIdx.x, tid = threadIdx.x;
  __shared__ float red[256];
  __shared__ float sc[T_];
  __shared__ float swE[H2_];

  // async-copy wE = alignW[0:512] into LDS (2 dwords per lane)
  {
    unsigned l0 = (unsigned)(size_t)(&swE[tid]);
    unsigned l1 = (unsigned)(size_t)(&swE[tid + 256]);
    const float* g0 = alignW + tid;
    const float* g1 = alignW + tid + 256;
    asm volatile("global_load_async_to_lds_b32 %0, %1, off"
                 :: "v"(l0), "v"(g0) : "memory");
    asm volatile("global_load_async_to_lds_b32 %0, %1, off"
                 :: "v"(l1), "v"(g1) : "memory");
    asm volatile("s_wait_asynccnt 0" ::: "memory");
  }

  // score[b,t] = enc[b,t]*wE + h1[b]*wS + kc*cov + kb + align_b
  float ps = 0.f, pkc = 0.f, pkb = 0.f;
  for (int j = tid; j < H2_; j += 256) {
    float w2 = alignW[2 * H2_ + j];
    ps  += h1[b * H2_ + j] * alignW[H2_ + j];
    pkc += covW[j] * w2;
    pkb += covB[j] * w2;
  }
  float a_s = block_reduce_sum(ps, red);     // barrier also covers async fill
  float kc  = block_reduce_sum(pkc, red);
  float kb  = block_reduce_sum(pkb, red);
  float base = a_s + kb + alignB[0];

  for (int t = tid; t < T_; t += 256) {
    const float* e = enc + ((size_t)b * T_ + t) * H2_;
    float d = 0.f;
    for (int j = 0; j < H2_; j += 4) {
      float4 ev = *(const float4*)(e + j);
      float4 wv = *(const float4*)(swE + j);
      d += ev.x * wv.x + ev.y * wv.y + ev.z * wv.z + ev.w * wv.w;
    }
    float cv = cov_in[b * T_ + t];
    sc[t] = tanhf(d + base + kc * cv);
  }
  __syncthreads();

  float m = -1e30f;
  for (int t = tid; t < T_; t += 256) m = fmaxf(m, sc[t]);
  m = block_reduce_max(m, red);

  float s = 0.f;
  for (int t = tid; t < T_; t += 256) { float e = __expf(sc[t] - m); sc[t] = e; s += e; }
  s = block_reduce_sum(s, red);
  const float inv = 1.0f / s;

  float lossp = 0.f;
  for (int t = tid; t < T_; t += 256) {
    float a  = sc[t] * inv;
    float cv = cov_in[b * T_ + t];
    attn_out[b * T_ + t] = a;
    new_cov[b * T_ + t]  = cv + a;
    lossp += fminf(a, cv);
    atomicAdd(&copy_prob[(size_t)b * V_ + src[b * T_ + t]], a);
  }
  lossp = block_reduce_sum(lossp, red);
  if (tid == 0) atomicAdd(cov_loss, lossp);
}

// context[b,h] = sum_t attn[b,t]*enc[b,t,h]; also builds fc1 input [s|context]
__global__ void context_kernel(const float* __restrict__ enc,
                               const float* __restrict__ attn,
                               const float* __restrict__ h1,
                               float* __restrict__ fc1_in) {  // [B, 1024]
  int b = blockIdx.x, h = threadIdx.x;                        // 64 x 512
  const float* e = enc + (size_t)b * T_ * H2_ + h;
  const float* a = attn + b * T_;
  float acc = 0.f;
  for (int t = 0; t < T_; ++t) acc += a[t] * e[(size_t)t * H2_];
  fc1_in[b * H4_ + H2_ + h] = acc;             // context
  fc1_in[b * H4_ + h]       = h1[b * H2_ + h]; // s
}

// p_gen = sigmoid([context, s, emb] . gen_W + gen_b)
__global__ void __launch_bounds__(256)
pgen_kernel(const float* __restrict__ fc1_in,   // [B,1024] = [s|context]
            const float* __restrict__ emb,      // [B,E]
            const float* __restrict__ genW,     // [1152]
            const float* __restrict__ genB,     // [1]
            float* __restrict__ pgen) {         // [B]
  int b = blockIdx.x, tid = threadIdx.x;
  __shared__ float red[256];
  float p = 0.f;
  for (int j = tid; j < H2_; j += 256) p += fc1_in[b * H4_ + H2_ + j] * genW[j];        // context
  for (int j = tid; j < H2_; j += 256) p += fc1_in[b * H4_ + j]       * genW[H2_ + j];  // s
  for (int j = tid; j < E_;  j += 256) p += emb[b * E_ + j]           * genW[H4_ + j];  // emb
  p = block_reduce_sum(p, red);
  if (tid == 0) pgen[b] = sigmoidf_(p + genB[0]);
}

// Vocab softmax + pointer-generator mix
__global__ void __launch_bounds__(256)
final_mix(const float* __restrict__ logits,     // [B,V]
          const float* __restrict__ copy_prob,  // [B,V]
          const float* __restrict__ pgen,       // [B]
          float* __restrict__ out) {            // [B,V]
  int b = blockIdx.x, tid = threadIdx.x;
  __shared__ float red[256];
  const float* L = logits + (size_t)b * V_;
  float m = -1e30f;
  for (int v = tid; v < V_; v += 256) m = fmaxf(m, L[v]);
  m = block_reduce_max(m, red);
  float s = 0.f;
  for (int v = tid; v < V_; v += 256) s += __expf(L[v] - m);
  s = block_reduce_sum(s, red);
  const float inv = 1.0f / s;
  const float pg = pgen[b], om = 1.0f - pg;
  for (int v = tid; v < V_; v += 256) {
    float P = __expf(L[v] - m) * inv;
    out[(size_t)b * V_ + v] = P * pg + copy_prob[(size_t)b * V_ + v] * om;
  }
}

// ---------------------------------------------------------------------------
// Launch
// ---------------------------------------------------------------------------
extern "C" void kernel_launch(void* const* d_in, const int* in_sizes, int n_in,
                              void* d_out, int out_size, void* d_ws, size_t ws_size,
                              hipStream_t stream) {
  (void)in_sizes; (void)n_in; (void)out_size; (void)ws_size;

  const int*   input_ids = (const int*)  d_in[0];
  const float* pre_h     = (const float*)d_in[1];   // [2,B,H2]
  const float* enc       = (const float*)d_in[2];   // [B,T,H2]
  const int*   src       = (const int*)  d_in[3];   // [B,T]
  const float* cov       = (const float*)d_in[4];   // [B,1,T]
  const float* emb_tab   = (const float*)d_in[5];
  const float* W_ih0     = (const float*)d_in[6];
  const float* W_hh0     = (const float*)d_in[7];
  const float* b_ih0     = (const float*)d_in[8];
  const float* b_hh0     = (const float*)d_in[9];
  const float* W_ih1     = (const float*)d_in[10];
  const float* W_hh1     = (const float*)d_in[11];
  const float* b_ih1     = (const float*)d_in[12];
  const float* b_hh1     = (const float*)d_in[13];
  const float* alignW    = (const float*)d_in[14];
  const float* alignB    = (const float*)d_in[15];
  const float* covW      = (const float*)d_in[16];
  const float* covB      = (const float*)d_in[17];
  const float* fc1_W     = (const float*)d_in[18];
  const float* fc1_b     = (const float*)d_in[19];
  const float* fc2_W     = (const float*)d_in[20];
  const float* fc2_b     = (const float*)d_in[21];
  const float* genW      = (const float*)d_in[22];
  const float* genB      = (const float*)d_in[23];

  // Workspace layout (floats)
  float* ws     = (float*)d_ws;
  float* emb    = ws;                 // 8192
  float* gi     = ws + 8192;          // 98304
  float* gh     = ws + 106496;        // 98304
  float* fc1_in = ws + 204800;        // 65536
  float* ni     = ws + 270336;        // 65536
  float* pgen   = ws + 335872;        // 64
  float* logits = ws + 335936;        // 3,200,000  (~13.6 MB total)

  // Output layout (floats, concat in return order)
  float* out       = (float*)d_out;              // [B,V]
  float* hidden    = out + (size_t)B_ * V_;      // [2,B,H2]
  float* attn      = hidden + 2 * B_ * H2_;      // [B,T]
  float* copy_prob = attn + B_ * T_;             // [B,V]
  float* new_cov   = copy_prob + (size_t)B_ * V_;// [B,T]
  float* cov_loss  = new_cov + B_ * T_;          // [1]
  float* h0 = hidden;
  float* h1 = hidden + B_ * H2_;

  // 0) zero the scatter targets (copy_prob + new_cov + loss are contiguous)
  {
    size_t n = (size_t)B_ * V_ + B_ * T_ + 1;
    zero_kernel<<<(unsigned)((n + 255) / 256), 256, 0, stream>>>(copy_prob, n);
  }

  // 1) embedding gather
  embed_kernel<<<B_, E_, 0, stream>>>(input_ids, emb_tab, emb);

  // 2) GRU cell 0  (96 col-tiles -> 12 blocks of 8 waves)
  wmma_gemm_xwt<0><<<12, 256, 0, stream>>>(emb,   W_ih0, b_ih0, gi, G3_, E_);
  wmma_gemm_xwt<0><<<12, 256, 0, stream>>>(pre_h, W_hh0, b_hh0, gh, G3_, H2_);
  gru_gate<<<(B_ * H2_) / 256, 256, 0, stream>>>(gi, gh, pre_h, h0);

  // 3) GRU cell 1
  wmma_gemm_xwt<0><<<12, 256, 0, stream>>>(h0,               W_ih1, b_ih1, gi, G3_, H2_);
  wmma_gemm_xwt<0><<<12, 256, 0, stream>>>(pre_h + B_ * H2_, W_hh1, b_hh1, gh, G3_, H2_);
  gru_gate<<<(B_ * H2_) / 256, 256, 0, stream>>>(gi, gh, pre_h + B_ * H2_, h1);

  // 4) attention (+ new_coverage, coverage_loss, copy_prob scatter)
  attention_kernel<<<B_, 256, 0, stream>>>(enc, h1, cov, alignW, alignB,
                                           covW, covB, src,
                                           attn, new_cov, cov_loss, copy_prob);

  // 5) context + fc1 input assembly
  context_kernel<<<B_, H2_, 0, stream>>>(enc, attn, h1, fc1_in);

  // 6) fc1: [B,1024] @ [1024,1024]^T, tanh   (64 tiles -> 8 blocks)
  wmma_gemm_xwt<1><<<8, 256, 0, stream>>>(fc1_in, fc1_W, fc1_b, ni, H4_, H4_);

  // 7) fc2: [B,1024] @ [50000,1024]^T        (3125 tiles -> 391 blocks)
  wmma_gemm_xwt<0><<<391, 256, 0, stream>>>(ni, fc2_W, fc2_b, logits, V_, H4_);

  // 8) p_gen
  pgen_kernel<<<B_, 256, 0, stream>>>(fc1_in, emb, genW, genB, pgen);

  // 9) vocab softmax + mix
  final_mix<<<B_, 256, 0, stream>>>(logits, copy_prob, pgen, out);
}
